// GraphTransformer_40200893890742
// MI455X (gfx1250) — matrix operand order repaired and checked by reading, not compiled, not run on hardware
//
#include <hip/hip_runtime.h>
#include <math.h>

// Problem constants (match reference)
#define NNODES 50000
#define NEDGES 800000
#define IN_CH  128
#define HID    96
#define OUT_CH 40
#define HEADS  4
#define HEAD_D 24
#define EDGE_D 8
#define LAYERS 3
#define LN_EPS 1e-5f
#define ATT_SCALE 0.20412414523193154f  // 1/sqrt(24)

typedef __attribute__((ext_vector_type(16))) _Float16 v16h;
typedef __attribute__((ext_vector_type(8)))  float    v8f;
typedef __attribute__((ext_vector_type(4)))  int      v4i;

#if __has_builtin(__builtin_amdgcn_global_load_async_to_lds_b128)
#define HAVE_ASYNC_LDS 1
typedef __attribute__((address_space(1))) v4i* gv4i_p;  // global int4*
typedef __attribute__((address_space(3))) v4i* lv4i_p;  // LDS int4*
#else
#define HAVE_ASYNC_LDS 0
#endif

// ---------------- monotonic float<->uint encoding for atomic max ----------------
__device__ __forceinline__ unsigned fenc(float f) {
  unsigned u = __float_as_uint(f);
  return u ^ ((unsigned)((int)u >> 31) | 0x80000000u);
}
__device__ __forceinline__ float fdec(unsigned e) {
  unsigned u = (e & 0x80000000u) ? (e ^ 0x80000000u) : ~e;
  return __uint_as_float(u);
}

// ---------------- weight pack: f32 [K,Nc] -> f16 WMMA B-fragment layout --------
// grid = (ntiles, K/32), block = 32. Lane layout (contiguous k order, same as A in wmma_gemm):
//   lanes 0-15 hold K=ks*32+0..15, lanes 16-31 hold K=ks*32+16..31; n = lane&15.
// Packed element index: ((ntile*(K/32) + ks)*32 + lane)*16 + i  (i = K within 16)
__global__ void pack_w(const float* __restrict__ W, _Float16* __restrict__ P, int Nc) {
  const int ntile = blockIdx.x;
  const int ks    = blockIdx.y;           // gridDim.y == K/32
  const int lane  = threadIdx.x & 31;
  const int n     = ntile * 16 + (lane & 15);
  const int kb    = ks * 32 + (lane >> 4) * 16;
  v16h frag;
#pragma unroll
  for (int i = 0; i < 16; ++i)
    frag[i] = (n < Nc) ? (_Float16)W[(size_t)(kb + i) * Nc + n] : (_Float16)0.0f;
  *(v16h*)(P + ((size_t)(ntile * gridDim.y + ks) * 32 + lane) * 16) = frag;
}

// ---------------- WMMA GEMM: C[M,Nc] = A[M,K] @ W[K,Nc] + bias -----------------
// Block = 4 waves; block stages an 80-row A tile in LDS (padded rows), each wave
// owns one 16-col N-tile and computes MW=5 m-subtiles, reusing its B fragment.
// A tile is staged with GLOBAL_LOAD_ASYNC_TO_LDS_B128 (ASYNCcnt) when available.
template <int K, int MW>
__global__ void wmma_gemm(const float* __restrict__ A, const _Float16* __restrict__ P,
                          const float* __restrict__ bias, float* __restrict__ C, int Nc) {
  constexpr int KSTEPS = K / 32;
  constexpr int ROWS   = MW * 16;
  constexpr int LDSW   = K + 4;           // pad: 16 lanes hit distinct banks
  __shared__ float As[ROWS * LDSW];

  const int lane = threadIdx.x;           // 0..31
  const int wv   = threadIdx.y;           // 0..3
  const int tid  = wv * 32 + lane;
  const int rowBase = blockIdx.x * ROWS;

  // A tile rows are contiguous in memory -> straight coalesced float4 copy.
  // Trip count is exact (ROWS*C4 % 128 == 0): loop is uniform, EXEC all-ones.
  const float4* src4 = (const float4*)(A + (size_t)rowBase * K);
  float4* dst4 = (float4*)As;
  constexpr int C4 = K / 4;
  for (int idx = tid; idx < ROWS * C4; idx += 128) {
    int r = idx / C4, c = idx - r * C4;
#if HAVE_ASYNC_LDS
    __builtin_amdgcn_global_load_async_to_lds_b128(
        (gv4i_p)(src4 + idx), (lv4i_p)(dst4 + (r * (C4 + 1) + c)), 0, 0);
#else
    dst4[r * (C4 + 1) + c] = src4[idx];
#endif
  }
#if HAVE_ASYNC_LDS
#if __has_builtin(__builtin_amdgcn_s_wait_asynccnt)
  __builtin_amdgcn_s_wait_asynccnt(0);
#else
  asm volatile("s_wait_asynccnt 0x0" ::: "memory");
#endif
#endif
  __syncthreads();

  const int ntile = blockIdx.y * 4 + wv;
  const int NT = (Nc + 15) >> 4;
  if (ntile >= NT) return;                // whole idle wave; after the only barrier

  const int mloc = lane & 15;
  const int half = lane >> 4;
  const int col  = ntile * 16 + mloc;
  const bool colok = col < Nc;

  v8f acc[MW] = {};
#pragma unroll
  for (int ks = 0; ks < KSTEPS; ++ks) {
    // B fragment: one contiguous 32B v16h per lane from the packed buffer.
    const v16h bf = *(const v16h*)(P + ((size_t)(ntile * KSTEPS + ks) * 32 + lane) * 16);
#pragma unroll
    for (int m = 0; m < MW; ++m) {
      // A fragment from LDS: element i -> K = ks*32 + half*16 + i
      const float* ar = As + (m * 16 + mloc) * LDSW + ks * 32;
      v16h af;
#pragma unroll
      for (int i = 0; i < 16; ++i) {
        int kk = half * 16 + i;
        af[i] = (_Float16)ar[kk];
      }
      acc[m] = __builtin_amdgcn_wmma_f32_16x16x32_f16(false, af, false, bf,
                                                      (short)0, acc[m], false, false);
    }
  }

  if (!colok) return;
  const float bv = bias[col];
#pragma unroll
  for (int m = 0; m < MW; ++m) {
#pragma unroll
    for (int r = 0; r < 8; ++r) {
      // C/D layout: VGPR r -> M = r + half*8, N = lane&15
      int orow = rowBase + m * 16 + r + half * 8;
      C[(size_t)orow * Nc + col] = acc[m][r] + bv;
    }
  }
}

// ---------------- zero scratch (agg | amax | den) ----------------
__global__ void zero_u32(unsigned* __restrict__ p, int n) {
  int i = blockIdx.x * blockDim.x + threadIdx.x;
  if (i < n) p[i] = 0u;
}

// ---------------- edge pass 1: alpha + segment max (per (edge, head)) ----------
// grid*block == NEDGES*HEADS exactly (3.2M = 12500*256): no tail guard.
__global__ void edge_pass1(const int* __restrict__ ei, const float* __restrict__ ea,
                           const float* __restrict__ q, const float* __restrict__ k,
                           const float* __restrict__ We, float* __restrict__ alpha,
                           unsigned* __restrict__ amax) {
  __shared__ float sWe[EDGE_D * HID];     // 3KB layer slice, broadcast-read
  for (int i = threadIdx.x; i < EDGE_D * HID; i += blockDim.x) sWe[i] = We[i];
  __syncthreads();

  const int tid = blockIdx.x * blockDim.x + threadIdx.x;
  const int e = tid >> 2;
  const int h = tid & 3;
  const int src = ei[e];
  const int dst = ei[NEDGES + e];

  float a8[EDGE_D];
#pragma unroll
  for (int j = 0; j < EDGE_D; ++j) a8[j] = ea[(size_t)e * EDGE_D + j];

  const float* qr = q + (size_t)dst * HID + h * HEAD_D;
  const float* kr = k + (size_t)src * HID + h * HEAD_D;
  const float* wc = sWe + h * HEAD_D;

  float s = 0.f;
#pragma unroll
  for (int d = 0; d < HEAD_D; ++d) {
    float ev = 0.f;  // fused edge embedding: e[h,d] = sum_j ea[j]*We[j, h*24+d]
#pragma unroll
    for (int j = 0; j < EDGE_D; ++j) ev = fmaf(a8[j], wc[j * HID + d], ev);
    s = fmaf(qr[d], kr[d] + ev, s);
  }
  s *= ATT_SCALE;
  alpha[tid] = s;
  atomicMax(&amax[dst * HEADS + h], fenc(s));
}

// ---------------- edge pass 2: exp, segment sums (den, unnormalized agg) -------
__global__ void edge_pass2(const int* __restrict__ ei, const float* __restrict__ ea,
                           const float* __restrict__ v, const float* __restrict__ We,
                           const float* __restrict__ alpha, const unsigned* __restrict__ amax,
                           float* __restrict__ den, float* __restrict__ agg) {
  __shared__ float sWe[EDGE_D * HID];
  for (int i = threadIdx.x; i < EDGE_D * HID; i += blockDim.x) sWe[i] = We[i];
  __syncthreads();

  const int tid = blockIdx.x * blockDim.x + threadIdx.x;
  const int e = tid >> 2;
  const int h = tid & 3;
  const int src = ei[e];
  const int dst = ei[NEDGES + e];

  const float ax = fdec(amax[dst * HEADS + h]);
  const float ex = expf(alpha[tid] - ax);
  atomicAdd(&den[dst * HEADS + h], ex);

  float a8[EDGE_D];
#pragma unroll
  for (int j = 0; j < EDGE_D; ++j) a8[j] = ea[(size_t)e * EDGE_D + j];

  const float* vr = v + (size_t)src * HID + h * HEAD_D;
  const float* wc = sWe + h * HEAD_D;
  float* ar = agg + (size_t)dst * HID + h * HEAD_D;
#pragma unroll
  for (int d = 0; d < HEAD_D; ++d) {
    float ev = 0.f;
#pragma unroll
    for (int j = 0; j < EDGE_D; ++j) ev = fmaf(a8[j], wc[j * HID + d], ev);
    atomicAdd(&ar[d], ex * (vr[d] + ev));
  }
}

// ---------------- per-node finalize: agg/den + skip, residual, LN, GELU --------
// One wave (32 lanes) per node; 3 channels per lane (96 = 3*32). wave32 shuffles.
__global__ void node_finalize(const float* __restrict__ hin, const float* __restrict__ agg,
                              const float* __restrict__ den, const float* __restrict__ skipb,
                              const float* __restrict__ ln_g, const float* __restrict__ ln_b,
                              float* __restrict__ hout) {
  const int wid  = threadIdx.x >> 5;
  const int lane = threadIdx.x & 31;
  const int node = blockIdx.x * (blockDim.x >> 5) + wid;
  if (node >= NNODES) return;

  float hv[3];
#pragma unroll
  for (int t = 0; t < 3; ++t) {
    int c = lane + 32 * t;
    int head = c / HEAD_D;
    float dn = den[node * HEADS + head];
    float attn = (dn > 0.f) ? agg[(size_t)node * HID + c] / dn : 0.f;
    // skipb already includes bskip (GEMM bias); out = attn + skip; h' = h + out
    hv[t] = hin[(size_t)node * HID + c] + attn + skipb[(size_t)node * HID + c];
  }
  float s = hv[0] + hv[1] + hv[2];
#pragma unroll
  for (int o = 16; o > 0; o >>= 1) s += __shfl_xor(s, o, 32);
  const float mu = s * (1.f / HID);
  float s2 = 0.f;
#pragma unroll
  for (int t = 0; t < 3; ++t) { float d = hv[t] - mu; s2 = fmaf(d, d, s2); }
#pragma unroll
  for (int o = 16; o > 0; o >>= 1) s2 += __shfl_xor(s2, o, 32);
  const float inv = rsqrtf(s2 * (1.f / HID) + LN_EPS);
#pragma unroll
  for (int t = 0; t < 3; ++t) {
    int c = lane + 32 * t;
    float xn = (hv[t] - mu) * inv * ln_g[c] + ln_b[c];
    // exact GELU (approximate=False): 0.5*x*(1+erf(x/sqrt(2)))
    float g = 0.5f * xn * (1.f + erff(xn * 0.70710678118654752f));
    hout[(size_t)node * HID + c] = g;
  }
}

// ---------------- per-node log_softmax over 40 logits --------------------------
__global__ void out_logsoftmax(const float* __restrict__ logits, float* __restrict__ out) {
  const int wid  = threadIdx.x >> 5;
  const int lane = threadIdx.x & 31;
  const int node = blockIdx.x * (blockDim.x >> 5) + wid;
  if (node >= NNODES) return;
  const float* row = logits + (size_t)node * OUT_CH;
  const bool hi = (lane + 32) < OUT_CH;  // lanes 0..7 carry elems 32..39
  float v0 = row[lane];
  float v1 = hi ? row[lane + 32] : -3.4e38f;
  float m = fmaxf(v0, v1);
#pragma unroll
  for (int o = 16; o > 0; o >>= 1) m = fmaxf(m, __shfl_xor(m, o, 32));
  float s = expf(v0 - m) + (hi ? expf(v1 - m) : 0.f);
#pragma unroll
  for (int o = 16; o > 0; o >>= 1) s += __shfl_xor(s, o, 32);
  const float lse = m + logf(s);
  out[(size_t)node * OUT_CH + lane] = v0 - lse;
  if (hi) out[(size_t)node * OUT_CH + lane + 32] = v1 - lse;
}

// ================================ host launcher ================================
extern "C" void kernel_launch(void* const* d_in, const int* in_sizes, int n_in,
                              void* d_out, int out_size, void* d_ws, size_t ws_size,
                              hipStream_t stream) {
  (void)in_sizes; (void)n_in; (void)out_size; (void)ws_size;
  const float* x      = (const float*)d_in[0];
  const int*   ei     = (const int*)d_in[1];     // [2,E] int32 (jax x64 off)
  const float* ea     = (const float*)d_in[2];   // [E,8]
  const float* W_in   = (const float*)d_in[3];   // [128,96]
  const float* b_in   = (const float*)d_in[4];
  const float* Wq     = (const float*)d_in[5];   // [L,96,96]
  const float* bq     = (const float*)d_in[6];
  const float* Wk     = (const float*)d_in[7];
  const float* bk     = (const float*)d_in[8];
  const float* Wv     = (const float*)d_in[9];
  const float* bv     = (const float*)d_in[10];
  const float* We     = (const float*)d_in[11];  // [L,8,96]
  const float* Wskip  = (const float*)d_in[12];
  const float* bskip  = (const float*)d_in[13];
  const float* ln_g   = (const float*)d_in[14];
  const float* ln_b   = (const float*)d_in[15];
  const float* W_out  = (const float*)d_in[16];  // [96,40]
  const float* b_out  = (const float*)d_in[17];
  float* out = (float*)d_out;

  // ---- workspace carve-up ----
  float* ws = (float*)d_ws;
  float* hA    = ws;                          // [N,96] current h
  float* hB    = hA   + (size_t)NNODES * HID; // [N,96] next h
  float* qbuf  = hB   + (size_t)NNODES * HID; // [N,96]
  float* kbuf  = qbuf + (size_t)NNODES * HID; // [N,96]
  float* vbuf  = kbuf + (size_t)NNODES * HID; // [N,96]
  float* skipb = vbuf + (size_t)NNODES * HID; // [N,96]
  float* agg   = skipb + (size_t)NNODES * HID; // [N,96] (zero region start)
  unsigned* amax = (unsigned*)(agg + (size_t)NNODES * HID); // [N,4]
  float* den   = (float*)amax + (size_t)NNODES * HEADS;     // [N,4]
  float* alpha = den + (size_t)NNODES * HEADS;              // [E,4]
  float* logits = qbuf;  // reuse q buffer after the last layer: [N,40]

  // packed f16 weights (32B aligned region after alpha)
  uintptr_t pb = (uintptr_t)(alpha + (size_t)NEDGES * HEADS);
  pb = (pb + 31) & ~(uintptr_t)31;
  _Float16* pWin  = (_Float16*)pb;                       // 6*4*512 = 12288
  _Float16* pWl   = pWin + 6 * 4 * 512;                  // [L][4] * 6*3*512
  _Float16* pWout = pWl + (size_t)LAYERS * 4 * 6 * 3 * 512; // 3*3*512

  const size_t PACK96 = 6 * 3 * 512;  // packed size of a 96x96 matrix

  // ---- pack all weights into WMMA B-fragment layout (once per launch) ----
  pack_w<<<dim3(6, IN_CH / 32), 32, 0, stream>>>(W_in, pWin, HID);
  for (int l = 0; l < LAYERS; ++l) {
    pack_w<<<dim3(6, HID / 32), 32, 0, stream>>>(Wq + (size_t)l * HID * HID, pWl + (l * 4 + 0) * PACK96, HID);
    pack_w<<<dim3(6, HID / 32), 32, 0, stream>>>(Wk + (size_t)l * HID * HID, pWl + (l * 4 + 1) * PACK96, HID);
    pack_w<<<dim3(6, HID / 32), 32, 0, stream>>>(Wv + (size_t)l * HID * HID, pWl + (l * 4 + 2) * PACK96, HID);
    pack_w<<<dim3(6, HID / 32), 32, 0, stream>>>(Wskip + (size_t)l * HID * HID, pWl + (l * 4 + 3) * PACK96, HID);
  }
  pack_w<<<dim3(3, HID / 32), 32, 0, stream>>>(W_out, pWout, OUT_CH);

  const dim3 gblk(32, 4);
  const int  MGRID = NNODES / 16 / 5;  // 625 exact (3125 m-tiles, MW=5)

  // h = x @ W_in + b_in
  wmma_gemm<IN_CH, 5><<<dim3(MGRID, 2), gblk, 0, stream>>>(x, pWin, b_in, hA, HID);

  const int edgeBlocks = (NEDGES * HEADS) / 256;       // 12500 exact
  const int zeroWords  = NNODES * (HID + 2 * HEADS);   // agg + amax + den
  const int nodeBlocks = (NNODES + 7) / 8;             // 8 waves per block

  for (int l = 0; l < LAYERS; ++l) {
    const float* We_l = We + (size_t)l * EDGE_D * HID;

    zero_u32<<<(zeroWords + 255) / 256, 256, 0, stream>>>((unsigned*)agg, zeroWords);

    wmma_gemm<HID, 5><<<dim3(MGRID, 2), gblk, 0, stream>>>(hA, pWl + (l * 4 + 0) * PACK96, bq + l * HID, qbuf, HID);
    wmma_gemm<HID, 5><<<dim3(MGRID, 2), gblk, 0, stream>>>(hA, pWl + (l * 4 + 1) * PACK96, bk + l * HID, kbuf, HID);
    wmma_gemm<HID, 5><<<dim3(MGRID, 2), gblk, 0, stream>>>(hA, pWl + (l * 4 + 2) * PACK96, bv + l * HID, vbuf, HID);
    wmma_gemm<HID, 5><<<dim3(MGRID, 2), gblk, 0, stream>>>(hA, pWl + (l * 4 + 3) * PACK96, bskip + l * HID, skipb, HID);

    edge_pass1<<<edgeBlocks, 256, 0, stream>>>(ei, ea, qbuf, kbuf, We_l, alpha, amax);
    edge_pass2<<<edgeBlocks, 256, 0, stream>>>(ei, ea, vbuf, We_l, alpha, amax, den, agg);

    node_finalize<<<nodeBlocks, 256, 0, stream>>>(hA, agg, den, skipb,
                                                  ln_g + l * HID, ln_b + l * HID, hB);
    float* t = hA; hA = hB; hB = t;
  }

  // logits = h @ W_out + b_out  (3 n-tiles; col guard masks 40 < 48)
  wmma_gemm<HID, 5><<<dim3(MGRID, 1), gblk, 0, stream>>>(hA, pWout, b_out, logits, OUT_CH);
  out_logsoftmax<<<nodeBlocks, 256, 0, stream>>>(logits, out);
}
